// WindowAttention_64716567216784
// MI455X (gfx1250) — matrix operand order repaired and checked by reading, not compiled, not run on hardware
//
#include <hip/hip_runtime.h>

// ---------------------------------------------------------------------------
// Window attention for MI455X (gfx1250, wave32, WMMA, async-to-LDS).
// fp32 inputs pre-converted to bf16 once; all matmuls v_wmma_f32_16x16x32_bf16
// with f32 accumulation; GEMM tiles double-buffered via async LDS copies.
// ---------------------------------------------------------------------------

typedef __attribute__((ext_vector_type(16))) __bf16 v16bf;
typedef __attribute__((ext_vector_type(8)))  float  v8f;

#define BQ   4096
#define NH   12
#define NT   49
#define HD   32
#define CDIM 384

// workspace layout (ushort elements)
#define OFF_XBF   0L            // x bf16 [200704,384]; reused as aout after qkv
#define OFF_Q     77070336L     // [4096,12,49,32]
#define OFF_K     154140672L    // [4096,12,49,32]
#define OFF_V     231211008L    // [4096,12,32,64]  (t padded 49->64)
#define OFF_WQKV  331874304L    // qkv_w^T bf16 [1152,384]
#define OFF_WPROJ 332316672L    // proj_w^T bf16 [384,384]
#define VPAD_N    23592960L     // 49152*32*15 padded v elements

__device__ __forceinline__ unsigned short f2bf(float f) {
    unsigned u = __float_as_uint(f);
    unsigned r = u + 0x7FFFu + ((u >> 16) & 1u);   // RNE
    return (unsigned short)(r >> 16);
}

// ---------------- async global -> LDS (gfx1250), with safe fallback --------
#if defined(__gfx1250__) && __has_builtin(__builtin_amdgcn_global_load_async_to_lds_b128)
#define HAVE_ASYNC_LDS 1
#else
#define HAVE_ASYNC_LDS 0
#endif

typedef __attribute__((__vector_size__(4 * sizeof(int)))) int v4i;
typedef __attribute__((address_space(1))) v4i* gv4i_p;   // global ptr to int4
typedef __attribute__((address_space(3))) v4i* lv4i_p;   // LDS ptr to int4

__device__ __forceinline__ void async_cp16(const void* g, void* l) {
#if HAVE_ASYNC_LDS
    __builtin_amdgcn_global_load_async_to_lds_b128((gv4i_p)g, (lv4i_p)l, 0, 0);
#else
    *(uint4*)l = *(const uint4*)g;
#endif
}
__device__ __forceinline__ void async_wait0() {
#if HAVE_ASYNC_LDS
#if __has_builtin(__builtin_amdgcn_s_wait_asynccnt)
    __builtin_amdgcn_s_wait_asynccnt(0);
#else
    asm volatile("s_wait_asynccnt 0" ::: "memory");
#endif
#endif
}
__device__ __forceinline__ void async_wait4() {
#if HAVE_ASYNC_LDS
#if __has_builtin(__builtin_amdgcn_s_wait_asynccnt)
    __builtin_amdgcn_s_wait_asynccnt(4);
#else
    asm volatile("s_wait_asynccnt 4" ::: "memory");
#endif
#endif
}

// ---------------- WMMA fragment helpers ------------------------------------
struct Bits256 { uint4 lo, hi; };

// A fragment: 16(M=lane%16) x 32(K contiguous); per ISA A-table
// lanes 0-15: K={0..7,16..23}; lanes 16-31: K={8..15,24..31}
__device__ __forceinline__ v16bf frag_ld_a(const unsigned short* base, int ld) {
    const int lane = threadIdx.x & 31;
    const int r    = lane & 15;
    const int h8   = (lane >> 4) << 3;
    const unsigned short* row = base + r * ld + h8;
    Bits256 b;
    b.lo = *(const uint4*)(row);
    b.hi = *(const uint4*)(row + 16);
    return __builtin_bit_cast(v16bf, b);
}
// B fragment from B^T [N=lane%16][K contiguous]; lanes0-15: K={0..7,16..23}, lanes16-31: K={8..15,24..31}
__device__ __forceinline__ v16bf frag_ld_b(const unsigned short* base, int ld) {
    const int lane = threadIdx.x & 31;
    const int r    = lane & 15;
    const int k0   = (lane >> 4) << 3;
    const unsigned short* p = base + r * ld + k0;
    Bits256 b;
    b.lo = *(const uint4*)(p);
    b.hi = *(const uint4*)(p + 16);
    return __builtin_bit_cast(v16bf, b);
}
__device__ __forceinline__ v8f wmma_bf16(v16bf a, v16bf b, v8f c) {
    return __builtin_amdgcn_wmma_f32_16x16x32_bf16(false, a, false, b, (short)0, c, false, false);
}

// ---------------------------------------------------------------------------
// Pre-conversion kernels (memory bound, run once).
// ---------------------------------------------------------------------------
__global__ __launch_bounds__(256) void convert_x(
    const float* __restrict__ X, unsigned short* __restrict__ Y, long n4)
{
    for (long i = (long)blockIdx.x * 256 + threadIdx.x; i < n4; i += (long)gridDim.x * 256) {
        const float4 f = *(const float4*)(X + i * 4);
        uint2 u;
        u.x = (unsigned)f2bf(f.x) | ((unsigned)f2bf(f.y) << 16);
        u.y = (unsigned)f2bf(f.z) | ((unsigned)f2bf(f.w) << 16);
        *(uint2*)(Y + i * 4) = u;
    }
}

// W [K][N] fp32 -> Wt [N][K] bf16 (K-contiguous for frag loaders)
__global__ __launch_bounds__(256) void convert_wT(
    const float* __restrict__ W, unsigned short* __restrict__ Wt, int K, int N)
{
    const int i = blockIdx.x * 256 + threadIdx.x;
    if (i < K * N) {
        const int n = i / K, k = i - n * K;
        Wt[i] = f2bf(W[(long)k * N + n]);
    }
}

// zero the t=49..63 padding of v [49152][32][64]
__global__ __launch_bounds__(256) void zero_vpad(unsigned short* __restrict__ v)
{
    const long i = (long)blockIdx.x * 256 + threadIdx.x;
    if (i < VPAD_N) {
        const long r = i / 15;
        const int  t = 49 + (int)(i - r * 15);
        v[r * 64 + t] = 0;
    }
}

// ---------------------------------------------------------------------------
// Shared GEMM tile stage: copy 128x32 bf16 tiles of A and B^T into LDS.
// 512 16-byte chunks each; 256 threads issue 2 A + 2 B async copies apiece.
// ---------------------------------------------------------------------------
__device__ __forceinline__ void gemm_stage(
    const unsigned short* __restrict__ Abase,
    const unsigned short* __restrict__ Bbase,
    int m0, int n0, int k0, int tid,
    unsigned short* As, unsigned short* Bs)
{
    #pragma unroll
    for (int p = 0; p < 2; ++p) {
        const int c   = tid + p * 256;     // 0..511
        const int row = c >> 2;
        const int cc  = (c & 3) * 8;
        async_cp16(Abase + (long)(m0 + row) * 384 + k0 + cc, As + row * 32 + cc);
        async_cp16(Bbase + (long)(n0 + row) * 384 + k0 + cc, Bs + row * 32 + cc);
    }
}

// ---------------------------------------------------------------------------
// Kernel 1: qkv = x @ qkv_w + b; scatter q (scaled), k to [b,h,49,32],
//           v transposed+padded to [b,h,32,64].  M=200704,N=1152,K=384.
// ---------------------------------------------------------------------------
__global__ __launch_bounds__(256) void qkv_gemm(
    const unsigned short* __restrict__ X,    // bf16 [200704,384]
    const unsigned short* __restrict__ Wt,   // bf16 [1152,384]
    const float* __restrict__ bias,          // [1152]
    unsigned short* __restrict__ q,
    unsigned short* __restrict__ k,
    unsigned short* __restrict__ v)
{
    __shared__ __attribute__((aligned(16))) unsigned short As[2][128 * 32];
    __shared__ __attribute__((aligned(16))) unsigned short Bs[2][128 * 32];

    const int tid  = threadIdx.x;
    const int m0   = blockIdx.y * 128;
    const int n0   = blockIdx.x * 128;
    const int wave = tid >> 5;
    const int wm   = (wave >> 2) * 64;
    const int wn   = (wave & 3) * 32;

    v8f zero = {};
    v8f acc[4][2];
    #pragma unroll
    for (int i = 0; i < 4; ++i)
        #pragma unroll
        for (int j = 0; j < 2; ++j) acc[i][j] = zero;

    gemm_stage(X, Wt, m0, n0, 0, tid, As[0], Bs[0]);

    for (int kt = 0; kt < 12; ++kt) {
        const int cur = kt & 1;
        if (kt < 11) {
            gemm_stage(X, Wt, m0, n0, (kt + 1) * 32, tid, As[cur ^ 1], Bs[cur ^ 1]);
            async_wait4();                 // previous tile landed; next in flight
        } else {
            async_wait0();
        }
        __syncthreads();

        v16bf bf[2];
        #pragma unroll
        for (int j = 0; j < 2; ++j) bf[j] = frag_ld_b(Bs[cur] + (wn + j * 16) * 32, 32);
        #pragma unroll
        for (int i = 0; i < 4; ++i) {
            v16bf af = frag_ld_a(As[cur] + (wm + i * 16) * 32, 32);
            #pragma unroll
            for (int j = 0; j < 2; ++j) acc[i][j] = wmma_bf16(af, bf[j], acc[i][j]);
        }
        __syncthreads();
    }

    const int lane = tid & 31;
    const int cn   = lane & 15;
    const int rb   = (lane >> 4) * 8;
    const float scale = 0.17677669529663687f;   // 1/sqrt(32)
    #pragma unroll
    for (int i = 0; i < 4; ++i) {
        #pragma unroll
        for (int j = 0; j < 2; ++j) {
            #pragma unroll
            for (int e = 0; e < 8; ++e) {
                const int m = m0 + wm + i * 16 + rb + e;
                const int n = n0 + wn + j * 16 + cn;
                float val = acc[i][j][e] + bias[n];
                const int sel = (n >= 768) ? 2 : (n >= 384 ? 1 : 0);
                const int nn  = n - sel * 384;
                const int h   = nn >> 5;
                const int d   = nn & 31;
                const int b   = m / 49;
                const int t   = m - b * 49;
                const long bh = (long)(b * 12 + h);
                if (sel == 0)      q[(bh * 49 + t) * 32 + d] = f2bf(val * scale);
                else if (sel == 1) k[(bh * 49 + t) * 32 + d] = f2bf(val);
                else               v[(bh * 32 + d) * 64 + t] = f2bf(val);
            }
        }
    }
}

// ---------------------------------------------------------------------------
// Kernel 2: per-(window,head) attention, tokens padded 49->64.
// ---------------------------------------------------------------------------
__global__ __launch_bounds__(128) void win_attn(
    const unsigned short* __restrict__ qg,   // [bh][49][32]
    const unsigned short* __restrict__ kg,   // [bh][49][32]
    const unsigned short* __restrict__ vg,   // [bh][32][64] pre-transposed, padded
    const float* __restrict__ bias_table,    // [169,12]
    unsigned short* __restrict__ aout)       // [4096,49,384] bf16
{
    __shared__ __attribute__((aligned(16))) unsigned short q_s[64 * 32];
    __shared__ __attribute__((aligned(16))) unsigned short k_s[64 * 32];
    __shared__ __attribute__((aligned(16))) unsigned short v_s[32 * 64];
    __shared__ __attribute__((aligned(16))) float          s_s[64 * 64];
    __shared__ __attribute__((aligned(16))) unsigned short p_s[64 * 64];

    const int bh  = blockIdx.x;
    const int h   = bh % 12;
    const int b   = bh / 12;
    const int tid = threadIdx.x;

    // Stage q/k (49x32 = 196 16B chunks each) and v (32x64 = 256 chunks) async.
    const unsigned short* gq = qg + (long)bh * NT * HD;
    const unsigned short* gk = kg + (long)bh * NT * HD;
    const unsigned short* gv = vg + (long)bh * HD * 64;
    for (int c = tid; c < 196; c += 128) async_cp16(gq + c * 8, q_s + c * 8);
    for (int c = tid; c < 196; c += 128) async_cp16(gk + c * 8, k_s + c * 8);
    for (int c = tid; c < 256; c += 128) async_cp16(gv + c * 8, v_s + c * 8);
    {   // zero q/k rows 49..63 (elements 1568..2047 = 60 uint4 each)
        const uint4 z = {0u, 0u, 0u, 0u};
        for (int c = tid; c < 60; c += 128) {
            *(uint4*)(q_s + 1568 + c * 8) = z;
            *(uint4*)(k_s + 1568 + c * 8) = z;
        }
    }
    async_wait0();
    __syncthreads();

    const int wave = tid >> 5;
    const int lane = tid & 31;
    const int cn   = lane & 15;
    const int rb   = (lane >> 4) * 8;

    // S = q @ k^T  (wave w owns rows 16w..16w+15)
    {
        v16bf qf = frag_ld_a(q_s + wave * 16 * 32, 32);
        #pragma unroll
        for (int nt = 0; nt < 4; ++nt) {
            v16bf kf = frag_ld_b(k_s + nt * 16 * 32, 32);
            v8f acc = {};
            acc = wmma_bf16(qf, kf, acc);
            #pragma unroll
            for (int e = 0; e < 8; ++e)
                s_s[(wave * 16 + rb + e) * 64 + nt * 16 + cn] = acc[e];
        }
    }
    __syncthreads();

    // bias + masked softmax, one row per thread
    if (tid < 64) {
        const int i = tid;
        if (i < NT) {
            const int ci = i / 7, wi = i % 7;
            float mx = -1e30f;
            for (int j = 0; j < NT; ++j) {
                const int cj = j / 7, wj = j % 7;
                const int ridx = (ci - cj + 6) * 13 + (wi - wj + 6);
                const float val = s_s[i * 64 + j] + bias_table[ridx * 12 + h];
                s_s[i * 64 + j] = val;
                mx = fmaxf(mx, val);
            }
            float sum = 0.f;
            for (int j = 0; j < NT; ++j) {
                const float e = __expf(s_s[i * 64 + j] - mx);
                s_s[i * 64 + j] = e;
                sum += e;
            }
            const float inv = 1.f / sum;
            for (int j = 0; j < NT; ++j)  p_s[i * 64 + j] = f2bf(s_s[i * 64 + j] * inv);
            for (int j = NT; j < 64; ++j) p_s[i * 64 + j] = 0;
        } else {
            for (int j = 0; j < 64; ++j)  p_s[i * 64 + j] = 0;
        }
    }
    __syncthreads();

    // O = P @ v
    #pragma unroll
    for (int nt = 0; nt < 2; ++nt) {
        v8f acc = {};
        #pragma unroll
        for (int kk = 0; kk < 2; ++kk) {
            v16bf pf = frag_ld_a(p_s + wave * 16 * 64 + kk * 32, 64);
            v16bf vf = frag_ld_b(v_s + nt   * 16 * 64 + kk * 32, 64);
            acc = wmma_bf16(pf, vf, acc);
        }
        #pragma unroll
        for (int e = 0; e < 8; ++e) {
            const int t = wave * 16 + rb + e;
            if (t < NT) {
                const int d = nt * 16 + cn;
                aout[((long)b * NT + t) * CDIM + h * 32 + d] = f2bf(acc[e]);
            }
        }
    }
}

// ---------------------------------------------------------------------------
// Kernel 3: out = attn_out @ proj_w + proj_b.  M=200704, N=384, K=384.
// ---------------------------------------------------------------------------
__global__ __launch_bounds__(256) void proj_gemm(
    const unsigned short* __restrict__ A,    // bf16 [200704,384]
    const unsigned short* __restrict__ Wt,   // bf16 [384,384] (transposed)
    const float* __restrict__ bias,          // [384]
    float* __restrict__ out)                 // fp32 [200704,384]
{
    __shared__ __attribute__((aligned(16))) unsigned short As[2][128 * 32];
    __shared__ __attribute__((aligned(16))) unsigned short Bs[2][128 * 32];

    const int tid  = threadIdx.x;
    const int m0   = blockIdx.y * 128;
    const int n0   = blockIdx.x * 128;
    const int wave = tid >> 5;
    const int wm   = (wave >> 2) * 64;
    const int wn   = (wave & 3) * 32;

    v8f zero = {};
    v8f acc[4][2];
    #pragma unroll
    for (int i = 0; i < 4; ++i)
        #pragma unroll
        for (int j = 0; j < 2; ++j) acc[i][j] = zero;

    gemm_stage(A, Wt, m0, n0, 0, tid, As[0], Bs[0]);

    for (int kt = 0; kt < 12; ++kt) {
        const int cur = kt & 1;
        if (kt < 11) {
            gemm_stage(A, Wt, m0, n0, (kt + 1) * 32, tid, As[cur ^ 1], Bs[cur ^ 1]);
            async_wait4();
        } else {
            async_wait0();
        }
        __syncthreads();

        v16bf bf[2];
        #pragma unroll
        for (int j = 0; j < 2; ++j) bf[j] = frag_ld_b(Bs[cur] + (wn + j * 16) * 32, 32);
        #pragma unroll
        for (int i = 0; i < 4; ++i) {
            v16bf af = frag_ld_a(As[cur] + (wm + i * 16) * 32, 32);
            #pragma unroll
            for (int j = 0; j < 2; ++j) acc[i][j] = wmma_bf16(af, bf[j], acc[i][j]);
        }
        __syncthreads();
    }

    const int lane = tid & 31;
    const int cn   = lane & 15;
    const int rb   = (lane >> 4) * 8;
    #pragma unroll
    for (int i = 0; i < 4; ++i) {
        #pragma unroll
        for (int j = 0; j < 2; ++j) {
            #pragma unroll
            for (int e = 0; e < 8; ++e) {
                const int m = m0 + wm + i * 16 + rb + e;
                const int n = n0 + wn + j * 16 + cn;
                out[(long)m * 384 + n] = acc[i][j][e] + bias[n];
            }
        }
    }
}

// ---------------------------------------------------------------------------
extern "C" void kernel_launch(void* const* d_in, const int* in_sizes, int n_in,
                              void* d_out, int out_size, void* d_ws, size_t ws_size,
                              hipStream_t stream) {
    const float* x          = (const float*)d_in[0];   // [4096,49,384]
    const float* qkv_w      = (const float*)d_in[1];   // [384,1152]
    const float* qkv_b      = (const float*)d_in[2];   // [1152]
    const float* proj_w     = (const float*)d_in[3];   // [384,384]
    const float* proj_b     = (const float*)d_in[4];   // [384]
    const float* bias_table = (const float*)d_in[5];   // [169,12]
    float* out = (float*)d_out;

    unsigned short* ws     = (unsigned short*)d_ws;
    unsigned short* xbf    = ws + OFF_XBF;     // reused as aout after qkv_gemm
    unsigned short* q_ws   = ws + OFF_Q;
    unsigned short* k_ws   = ws + OFF_K;
    unsigned short* v_ws   = ws + OFF_V;
    unsigned short* wqkvT  = ws + OFF_WQKV;
    unsigned short* wprojT = ws + OFF_WPROJ;
    unsigned short* aout   = xbf;              // alias: x_bf16 dead after qkv_gemm

    convert_x <<<8192, 256, 0, stream>>>(x, xbf, 19267584L);            // 77M/4
    convert_wT<<<(442368 + 255) / 256, 256, 0, stream>>>(qkv_w,  wqkvT,  384, 1152);
    convert_wT<<<(147456 + 255) / 256, 256, 0, stream>>>(proj_w, wprojT, 384, 384);
    zero_vpad <<<(int)((VPAD_N + 255) / 256), 256, 0, stream>>>(v_ws);

    qkv_gemm <<<dim3(9, 1568), 256, 0, stream>>>(xbf, wqkvT, qkv_b, q_ws, k_ws, v_ws);
    win_attn <<<dim3(BQ * NH), 128, 0, stream>>>(q_ws, k_ws, v_ws, bias_table, aout);
    proj_gemm<<<dim3(3, 1568), 256, 0, stream>>>(aout, wprojT, proj_b, out);
}
